// DeformableCrossAttention_36584531427680
// MI455X (gfx1250) — matrix operand order repaired and checked by reading, not compiled, not run on hardware
//
#include <hip/hip_runtime.h>
#include <hip/hip_bf16.h>

// ---------------------------------------------------------------------------
// Types
// ---------------------------------------------------------------------------
typedef __attribute__((ext_vector_type(16))) __bf16          v16bf;
typedef __attribute__((ext_vector_type(8)))  float           v8f;
typedef __attribute__((ext_vector_type(8)))  unsigned short  us8;
typedef __attribute__((ext_vector_type(4)))  unsigned short  us4;
typedef __attribute__((ext_vector_type(4)))  float           f4;

union Frag {
  v16bf bf;
  us8   h[2];
  unsigned short u[16];
};

__device__ __forceinline__ unsigned short f2bf(float f) {
  unsigned u = __float_as_uint(f);
  u += 0x7FFFu + ((u >> 16) & 1u);          // round-to-nearest-even
  return (unsigned short)(u >> 16);
}

// ---------------------------------------------------------------------------
// Elementwise fp32 -> bf16 (4 elements / thread, b128 load, b64 store)
// ---------------------------------------------------------------------------
__global__ void convert_bf16(const float* __restrict__ in,
                             unsigned short* __restrict__ out) {
  const size_t i = (size_t)(blockIdx.x * 256 + threadIdx.x) * 4;
  const f4 v = *(const f4*)(in + i);
  us4 o;
  #pragma unroll
  for (int j = 0; j < 4; ++j) o[j] = f2bf(v[j]);
  *(us4*)(out + i) = o;
}

// ---------------------------------------------------------------------------
// Weight transpose + fp32 -> bf16 convert:  Wt[n*K + k] = bf16(W[k*N + n])
// ---------------------------------------------------------------------------
__global__ void wt_convert(const float* __restrict__ W,
                           unsigned short* __restrict__ Wt,
                           int K, int N) {
  int idx = blockIdx.x * 256 + threadIdx.x;   // over N*K (multiple of 256)
  int n = idx / K;
  int k = idx - n * K;
  Wt[idx] = f2bf(W[(size_t)k * N + n]);
}

// ---------------------------------------------------------------------------
// WMMA GEMM:  C[m][n] = sum_k A[m][k] * Bt[n][k]   (bf16 A and Bt, K%32==0)
//   Block = 256 threads = 8 waves (2 in M x 4 in N).
//   Wave tile = 32(M) x NT*16(N); block tile = 64(M) x NT*64(N).
//   EPI: 0 = scatter v-layout f32 (no bias)
//        1 = +bias, exact GELU, store bf16
//        2 = +bias, tanh, store f32
//        3 = +bias, store f32
// ---------------------------------------------------------------------------
template<int EPI, int NT>
__global__ __launch_bounds__(256)
void gemm_wmma(const unsigned short* __restrict__ A,
               const unsigned short* __restrict__ Bt,
               const float* __restrict__ bias,
               void* __restrict__ out,
               int M, int N, int K) {
  const int tid  = threadIdx.x;
  const int lane = tid & 31;
  const int wave = tid >> 5;        // 0..7
  const int wm   = wave >> 2;       // 0..1  (M direction)
  const int wn   = wave & 3;        // 0..3  (N direction)
  const int m0   = blockIdx.y * 64 + wm * 32;
  const int n0   = (blockIdx.x * 4 + wn) * (NT * 16);

  const int lr = lane & 15;         // row (A) / col (B,C) within 16-tile
  const int hi = lane >> 4;         // half-wave selector

  v8f acc[2][NT];
  #pragma unroll
  for (int i = 0; i < 2; ++i)
    #pragma unroll
    for (int j = 0; j < NT; ++j)
      #pragma unroll
      for (int r = 0; r < 8; ++r) acc[i][j][r] = 0.0f;

  // Per-lane base pointers (K advances contiguously).
  // A frag (16-bit A 16x32): lane<16 -> K {0..7} U {16..23}; lane>=16 -> +8.
  const unsigned short* apc[2];
  #pragma unroll
  for (int mi = 0; mi < 2; ++mi)
    apc[mi] = A + (size_t)(m0 + mi * 16 + lr) * K + hi * 8;
  // B frag (16-bit B 32x16): lane<16 -> K {0..7} U {16..23}; lane>=16 -> +8; col=lane&15
  const unsigned short* bpc[NT];
  #pragma unroll
  for (int ni = 0; ni < NT; ++ni)
    bpc[ni] = Bt + (size_t)(n0 + ni * 16 + lr) * K + hi * 8;

  for (int k0 = 0; k0 < K; k0 += 32) {
    Frag a[2], b[NT];
    #pragma unroll
    for (int mi = 0; mi < 2; ++mi) {
      a[mi].h[0] = *(const us8*)(apc[mi] + k0);
      a[mi].h[1] = *(const us8*)(apc[mi] + k0 + 16);
    }
    #pragma unroll
    for (int ni = 0; ni < NT; ++ni) {
      b[ni].h[0] = *(const us8*)(bpc[ni] + k0);
      b[ni].h[1] = *(const us8*)(bpc[ni] + k0 + 16);
    }
    #pragma unroll
    for (int mi = 0; mi < 2; ++mi)
      #pragma unroll
      for (int ni = 0; ni < NT; ++ni)
        acc[mi][ni] = __builtin_amdgcn_wmma_f32_16x16x32_bf16(
            false, a[mi].bf, false, b[ni].bf, (short)0, acc[mi][ni],
            false, false);
  }

  // Epilogue. C/D layout: element r -> row = 8*hi + r, col = lane&15.
  #pragma unroll
  for (int mi = 0; mi < 2; ++mi) {
    #pragma unroll
    for (int ni = 0; ni < NT; ++ni) {
      const int col = n0 + ni * 16 + lr;
      float bv = 0.0f;
      if (EPI != 0) bv = bias[col];
      #pragma unroll
      for (int r = 0; r < 8; ++r) {
        const int row = m0 + mi * 16 + hi * 8 + r;
        float c = acc[mi][ni][r];
        if (EPI == 0) {
          // v scatter: row = b*4096 + s, col = h*64 + d
          // -> v_flat[((b*16+h)*4096 + s)*64 + d]
          const int bb = row >> 12, s = row & 4095;
          const int h  = col >> 6,  d = col & 63;
          ((float*)out)[(((size_t)(bb * 16 + h) * 4096 + s) << 6) + d] = c;
        } else if (EPI == 1) {
          c += bv;
          const float g = 0.5f * c * (1.0f + erff(c * 0.70710678118654752f));
          ((unsigned short*)out)[(size_t)row * N + col] = f2bf(g);
        } else if (EPI == 2) {
          c += bv;
          ((float*)out)[(size_t)row * N + col] = tanhf(c);
        } else {
          c += bv;
          ((float*)out)[(size_t)row * N + col] = c;
        }
      }
    }
  }
}

// ---------------------------------------------------------------------------
// Deformable bilinear sampling + softmax(P) + weighted sum over P.
//   64 threads per (b', q, h'); thread -> channel d. Output bf16 (ushort)
//   inner[(b'*256+q)*1024 + h'*64 + d] for the final WMMA GEMM.
//   Reference permutation: loc row used = q*8 + b'; aw row = b'*256 + q.
// ---------------------------------------------------------------------------
__global__ void deform_sample(const float* __restrict__ v,      // (B*H,64,64,64)
                              const float* __restrict__ loc,    // (2048,256) tanh'd
                              const float* __restrict__ awl,    // (2048,128) logits
                              unsigned short* __restrict__ inner) {
  const int t = blockIdx.x * 256 + threadIdx.x;
  const int d = t & 63;
  const int g = t >> 6;           // (b*256+q)*16 + h, 32768 total
  const int h = g & 15;
  const int m = g >> 4;           // b*256 + q
  const int b = m >> 8;
  const int q = m & 255;
  const int locrow = q * 8 + b;

  // softmax over P=8 attention logits
  const float* ap = awl + (size_t)m * 128 + h * 8;
  float l[8];
  float mx = -3.0e38f;
  #pragma unroll
  for (int p = 0; p < 8; ++p) { l[p] = ap[p]; mx = fmaxf(mx, l[p]); }
  float s = 0.0f;
  #pragma unroll
  for (int p = 0; p < 8; ++p) { l[p] = __expf(l[p] - mx); s += l[p]; }
  const float inv = 1.0f / s;

  const float* lp = loc + (size_t)locrow * 256 + h * 16;
  const float* vb = v + (size_t)(b * 16 + h) * (4096 * 64);

  float acc = 0.0f;
  #pragma unroll
  for (int p = 0; p < 8; ++p) {
    const float px = (lp[p * 2 + 0] + 1.0f) * 0.5f * 63.0f;
    const float py = (lp[p * 2 + 1] + 1.0f) * 0.5f * 63.0f;
    const float x0f = floorf(px), y0f = floorf(py);
    const float wx = px - x0f, wy = py - y0f;
    const int x0 = (int)x0f, y0 = (int)y0f;
    float cv[4];
    #pragma unroll
    for (int cy = 0; cy < 2; ++cy) {
      #pragma unroll
      for (int cx = 0; cx < 2; ++cx) {
        const int yy = y0 + cy, xx = x0 + cx;
        const bool valid = (yy >= 0) & (yy < 64) & (xx >= 0) & (xx < 64);
        const int yc = min(max(yy, 0), 63);
        const int xc = min(max(xx, 0), 63);
        const float val = vb[(((size_t)yc << 6) + xc) * 64 + d];
        cv[cy * 2 + cx] = valid ? val : 0.0f;
      }
    }
    const float top = cv[0] * (1.0f - wx) + cv[1] * wx;
    const float bot = cv[2] * (1.0f - wx) + cv[3] * wx;
    acc += (l[p] * inv) * (top * (1.0f - wy) + bot * wy);
  }
  inner[(size_t)m * 1024 + h * 64 + d] = f2bf(acc);
}

// ---------------------------------------------------------------------------
// Host-side launcher
// ---------------------------------------------------------------------------
extern "C" void kernel_launch(void* const* d_in, const int* in_sizes, int n_in,
                              void* d_out, int out_size, void* d_ws, size_t ws_size,
                              hipStream_t stream) {
  (void)in_sizes; (void)n_in; (void)out_size; (void)ws_size;

  const float* x     = (const float*)d_in[0];   // (8,256,1024)
  const float* ctx   = (const float*)d_in[1];   // (8,4096,1024)
  // d_in[2] = Wq  (result unused by reference -> skipped)
  const float* Wv    = (const float*)d_in[3];   // (1024,1024)
  const float* Woff1 = (const float*)d_in[4];   // (1024,1024)
  const float* boff1 = (const float*)d_in[5];   // (1024,)
  const float* Woff2 = (const float*)d_in[6];   // (1024,256)
  const float* boff2 = (const float*)d_in[7];   // (256,)
  const float* Watt1 = (const float*)d_in[8];   // (1024,1024)
  const float* batt1 = (const float*)d_in[9];   // (1024,)
  const float* Watt2 = (const float*)d_in[10];  // (1024,128)
  const float* batt2 = (const float*)d_in[11];  // (128,)
  const float* Wout  = (const float*)d_in[12];  // (1024,1024)
  const float* bout  = (const float*)d_in[13];  // (1024,)

  // Workspace carve-out (256B aligned)
  char* w = (char*)d_ws;
  auto carve = [&](size_t bytes) -> void* {
    void* p = (void*)w;
    w += (bytes + 255) & ~((size_t)255);
    return p;
  };
  unsigned short* WvT    = (unsigned short*)carve((size_t)1024 * 1024 * 2);
  unsigned short* Woff1T = (unsigned short*)carve((size_t)1024 * 1024 * 2);
  unsigned short* Woff2T = (unsigned short*)carve((size_t)1024 * 256 * 2);
  unsigned short* Watt1T = (unsigned short*)carve((size_t)1024 * 1024 * 2);
  unsigned short* Watt2T = (unsigned short*)carve((size_t)1024 * 128 * 2);
  unsigned short* WoutT  = (unsigned short*)carve((size_t)1024 * 1024 * 2);
  unsigned short* ctxB   = (unsigned short*)carve((size_t)32768 * 1024 * 2);  // 64 MB
  unsigned short* xB     = (unsigned short*)carve((size_t)2048 * 1024 * 2);
  float*          vflat  = (float*)carve((size_t)128 * 4096 * 64 * 4);        // 128 MB
  unsigned short* h1     = (unsigned short*)carve((size_t)2048 * 1024 * 2);
  unsigned short* h2     = (unsigned short*)carve((size_t)2048 * 1024 * 2);
  float*          locb   = (float*)carve((size_t)2048 * 256 * 4);
  float*          awlb   = (float*)carve((size_t)2048 * 128 * 4);
  unsigned short* inner  = (unsigned short*)carve((size_t)2048 * 1024 * 2);

  const dim3 blk(256);

  // 1) activations fp32 -> bf16 (once, coalesced)
  convert_bf16<<<(32768 * 1024) / (256 * 4), blk, 0, stream>>>(ctx, ctxB);
  convert_bf16<<<(2048 * 1024) / (256 * 4), blk, 0, stream>>>(x, xB);

  // 2) weight transpose+convert (Wt[n][k] = bf16 W[k][n])
  wt_convert<<<(1024 * 1024) / 256, blk, 0, stream>>>(Wv,    WvT,    1024, 1024);
  wt_convert<<<(1024 * 1024) / 256, blk, 0, stream>>>(Woff1, Woff1T, 1024, 1024);
  wt_convert<<<(1024 * 256)  / 256, blk, 0, stream>>>(Woff2, Woff2T, 1024, 256);
  wt_convert<<<(1024 * 1024) / 256, blk, 0, stream>>>(Watt1, Watt1T, 1024, 1024);
  wt_convert<<<(1024 * 128)  / 256, blk, 0, stream>>>(Watt2, Watt2T, 1024, 128);
  wt_convert<<<(1024 * 1024) / 256, blk, 0, stream>>>(Wout,  WoutT,  1024, 1024);

  // 3) v = context @ Wv, scattered into (B*H, 64, 64, 64) sampling layout
  gemm_wmma<0, 4><<<dim3(1024 / 256, 32768 / 64), blk, 0, stream>>>(
      ctxB, WvT, nullptr, vflat, 32768, 1024, 1024);

  // 4) hidden activations: gelu(x@W + b) -> bf16
  gemm_wmma<1, 4><<<dim3(1024 / 256, 2048 / 64), blk, 0, stream>>>(
      xB, Woff1T, boff1, h1, 2048, 1024, 1024);
  gemm_wmma<1, 4><<<dim3(1024 / 256, 2048 / 64), blk, 0, stream>>>(
      xB, Watt1T, batt1, h2, 2048, 1024, 1024);

  // 5) loc = tanh(h1@W_off2 + b_off2); aw logits = h2@W_att2 + b_att2
  gemm_wmma<2, 4><<<dim3(256 / 256, 2048 / 64), blk, 0, stream>>>(
      h1, Woff2T, boff2, locb, 2048, 256, 1024);
  gemm_wmma<3, 2><<<dim3(128 / 128, 2048 / 64), blk, 0, stream>>>(
      h2, Watt2T, batt2, awlb, 2048, 128, 1024);

  // 6) bilinear sampling + softmax-weighted sum over P -> bf16 inner
  deform_sample<<<(32768 * 64) / 256, blk, 0, stream>>>(vflat, locb, awlb, inner);

  // 7) out = inner @ W_out + b_out -> fp32 d_out
  gemm_wmma<3, 4><<<dim3(1024 / 256, 2048 / 64), blk, 0, stream>>>(
      inner, WoutT, bout, (float*)d_out, 2048, 1024, 1024);
}
